// DeformConv_55705725829127
// MI455X (gfx1250) — matrix operand order repaired and checked by reading, not compiled, not run on hardware
//
#include <hip/hip_runtime.h>
#include <hip/hip_bf16.h>
#include <stdint.h>

#define BN_ 8
#define CI_ 256
#define CO_ 256
#define HH 64
#define WW 64
#define HW 4096
#define KK 9
#define CK 2304              // CI*K, K-dim order: k*CI + ci
#define NCHUNK 72            // CK/32

typedef __attribute__((ext_vector_type(16))) __bf16 v16bf;
typedef __attribute__((ext_vector_type(8)))  float  v8f;

union Frag16 { v16bf bf; unsigned u[8]; };

__device__ __forceinline__ unsigned bfpack(float a, float b) {
    unsigned ua = __float_as_uint(a), ub = __float_as_uint(b);
    ua = (ua + 0x7FFFu + ((ua >> 16) & 1u)) >> 16;
    ub = (ub + 0x7FFFu + ((ub >> 16) & 1u)) >> 16;
    return ua | (ub << 16);
}

__device__ __forceinline__ float bilin(const float* __restrict__ xp,
                                       const float4& w, const int4& i) {
    return xp[i.x] * w.x + xp[i.y] * w.y + xp[i.z] * w.z + xp[i.w] * w.w;
}

// Async global->LDS copy of 64 bytes per lane (4 x b128).  Per the CDNA5 ISA,
// the instruction offset is added to BOTH the LDS and global addresses.
__device__ __forceinline__ void async_copy64(unsigned lds_addr, const void* gaddr) {
    asm volatile(
        "global_load_async_to_lds_b128 %0, %1, off\n\t"
        "global_load_async_to_lds_b128 %0, %1, off offset:16\n\t"
        "global_load_async_to_lds_b128 %0, %1, off offset:32\n\t"
        "global_load_async_to_lds_b128 %0, %1, off offset:48"
        :: "v"(lds_addr), "v"(gaddr) : "memory");
}
__device__ __forceinline__ void wait_async0() {
    asm volatile("s_wait_asynccnt 0x0" ::: "memory");
}

// ---------------------------------------------------------------------------
// Kernel 1: pack w_dcn [CO][CI][3][3] fp32 -> bf16 WMMA B-fragments.
// B tile = 32(K) x 16(co).  Storage: wsB[(kc*16+ct)*32 + lane][v], v=0..7.
// Lanes 0-15: n=lane, VGPR v holds K={2v,2v+1};
// Lanes 16-31: n=lane-16, VGPR v holds K={16+2v,17+2v}.
// Global K index = kc*32 + klocal ;  k9 = K/CI, ci = K%CI.
// ---------------------------------------------------------------------------
__global__ __launch_bounds__(256) void pack_wdcn_kernel(
    const float* __restrict__ w_dcn, unsigned* __restrict__ wsB) {
    int gid  = blockIdx.x * 256 + threadIdx.x;     // 72*16*32 = 36864 total
    int lane = gid & 31;
    int tile = gid >> 5;                           // kc*16 + ct
    if (tile >= NCHUNK * 16) return;
    int ct = tile & 15, kc = tile >> 4;
    int n  = lane & 15;
    int co = ct * 16 + n;
    int koff = (lane & 16) ? 16 : 0;
    unsigned* dst = wsB + ((long)tile * 32 + lane) * 8;
#pragma unroll
    for (int v = 0; v < 8; ++v) {
        int klo = kc * 32 + koff + 2 * v;
        int ci0 = klo & 255, k90 = klo >> 8;       // CI_ = 256
        int ci1 = (klo + 1) & 255, k91 = (klo + 1) >> 8;
        float f0 = w_dcn[co * CK + ci0 * KK + k90];
        float f1 = w_dcn[co * CK + ci1 * KK + k91];
        dst[v] = bfpack(f0, f1);
    }
}

// ---------------------------------------------------------------------------
// Kernel 2: fold BN: scale = gamma*rsqrt(var+eps); bias = beta+(b_dcn-mean)*scale
// ---------------------------------------------------------------------------
__global__ __launch_bounds__(256) void bn_fold_kernel(
    const float* __restrict__ b_dcn, const float* __restrict__ gamma,
    const float* __restrict__ beta,  const float* __restrict__ rmean,
    const float* __restrict__ rvar,  float* __restrict__ scl,
    float* __restrict__ bia) {
    int c = threadIdx.x;
    float inv = gamma[c] * rsqrtf(rvar[c] + 1e-5f);
    scl[c] = inv;
    bia[c] = beta[c] + (b_dcn[c] - rmean[c]) * inv;
}

// ---------------------------------------------------------------------------
// Kernel 3: offset conv (27 ch, 3x3, direct) + sigmoid mask + bilinear
// weight/index precompute.  Block = 4 rows x 64 cols of one image.
// Output per (b,k,pixel): float4 corner weights (mask folded) + int4 indices.
// ---------------------------------------------------------------------------
__global__ __launch_bounds__(256) void offset_kernel(
    const float* __restrict__ x, const float* __restrict__ w_off,
    const float* __restrict__ b_off,
    float* __restrict__ wsW, int* __restrict__ wsI) {
    int blk  = blockIdx.x;                 // 0..127
    int b    = blk >> 4;                   // 16 blocks per image
    int tile = blk & 15;
    int t    = threadIdx.x;
    int row  = tile * 4 + (t >> 6);
    int col  = t & 63;

    float acc[27];
#pragma unroll
    for (int oc = 0; oc < 27; ++oc) acc[oc] = 0.f;

    __shared__ float wsh[8 * 27 * 9];      // 7776 B
    const float* xb = x + (long)b * CI_ * HW;

    for (int cc = 0; cc < CI_; cc += 8) {
        __syncthreads();
        for (int i = t; i < 8 * 243; i += 256) {
            int cl = i / 243, rem = i - cl * 243;
            int oc = rem / 9, kk = rem - oc * 9;
            wsh[i] = w_off[oc * (CI_ * 9) + (cc + cl) * 9 + kk];
        }
        __syncthreads();
#pragma unroll 1
        for (int cl = 0; cl < 8; ++cl) {
            const float* xp = xb + (long)(cc + cl) * HW;
            float xv[9];
#pragma unroll
            for (int dy = -1; dy <= 1; ++dy)
#pragma unroll
                for (int dx = -1; dx <= 1; ++dx) {
                    int yy = row + dy, xx = col + dx;
                    bool ok = (yy >= 0) & (yy < HH) & (xx >= 0) & (xx < WW);
                    xv[(dy + 1) * 3 + (dx + 1)] = ok ? xp[yy * WW + xx] : 0.f;
                }
            const float* wp = &wsh[cl * 243];
#pragma unroll
            for (int oc = 0; oc < 27; ++oc) {
                float s = acc[oc];
#pragma unroll
                for (int kk = 0; kk < 9; ++kk)
                    s = fmaf(xv[kk], wp[oc * 9 + kk], s);
                acc[oc] = s;
            }
        }
    }

    int pix = row * WW + col;
#pragma unroll
    for (int k = 0; k < 9; ++k) {
        int ky = k / 3 - 1, kx = k % 3 - 1;
        float oy = acc[2 * k]     + b_off[2 * k];
        float ox = acc[2 * k + 1] + b_off[2 * k + 1];
        float mk = acc[18 + k]    + b_off[18 + k];
        float mask = 1.f / (1.f + __expf(-mk));
        float py = (float)(row + ky) + oy;
        float px = (float)(col + kx) + ox;
        float y0f = floorf(py), x0f = floorf(px);
        float wy1 = py - y0f, wy0 = 1.f - wy1;
        float wx1 = px - x0f, wx0 = 1.f - wx1;
        int y0 = (int)y0f, x0 = (int)x0f;
        int y1 = y0 + 1,  x1 = x0 + 1;
        bool vy0 = (y0 >= 0) & (y0 < HH), vy1 = (y1 >= 0) & (y1 < HH);
        bool vx0 = (x0 >= 0) & (x0 < WW), vx1 = (x1 >= 0) & (x1 < WW);
        int cy0 = min(max(y0, 0), HH - 1), cy1 = min(max(y1, 0), HH - 1);
        int cx0 = min(max(x0, 0), WW - 1), cx1 = min(max(x1, 0), WW - 1);
        float4 w4;
        w4.x = (vy0 & vx0) ? wy0 * wx0 * mask : 0.f;
        w4.y = (vy0 & vx1) ? wy0 * wx1 * mask : 0.f;
        w4.z = (vy1 & vx0) ? wy1 * wx0 * mask : 0.f;
        w4.w = (vy1 & vx1) ? wy1 * wx1 * mask : 0.f;
        int4 i4;
        i4.x = cy0 * WW + cx0;  i4.y = cy0 * WW + cx1;
        i4.z = cy1 * WW + cx0;  i4.w = cy1 * WW + cx1;
        int sidx = (((b * 9 + k) << 12) + pix) << 2;   // *4096*4
        *(float4*)(wsW + sidx) = w4;
        *(int4*)(wsI + sidx)   = i4;
    }
}

// ---------------------------------------------------------------------------
// Kernel 4: fused bilinear-sample -> bf16 -> WMMA GEMM -> BN -> ReLU.
// Block = one image row (64 pixels) x all 256 co.  8 wave32 per block:
// wave (wn,wm): 16 pixels (wn) x 128 co (wm).  K loop: 72 chunks of 32.
//
// Double-buffered LDS pipeline (one __syncthreads per chunk):
//  - B: full 32x256 bf16 chunk (16 tiles = 16KB) copied with
//       global_load_async_to_lds_b128 (ASYNCcnt); each wave reads its 8
//       tiles into 8 distinct frags -> 8 back-to-back WMMAs
//       (sched_group_barrier pins DS-reads before the WMMA burst).
//  - A: fragment build split across wm halves (32 gathers/wave instead of
//       64), exchanged via LDS.
// Iteration kc reads buffers [kc&1], builds/writes buffers [(kc+1)&1];
// each wave waits its own ASYNCcnt to 0 before the next barrier.
// ---------------------------------------------------------------------------
__global__ __launch_bounds__(256) void dcn_wmma_kernel(
    const float* __restrict__ x,   const float* __restrict__ wsW,
    const int* __restrict__ wsI,   const unsigned* __restrict__ wsB,
    const float* __restrict__ scl, const float* __restrict__ bia,
    float* __restrict__ out) {
    __shared__ unsigned ldsB[2][4096];       // 32 KB: [tile(16)][lane(32)][8]
    __shared__ unsigned ldsA[2][4][32][8];   //  8 KB: [wn][lane][8]

    int blk  = blockIdx.x;            // 0..511 = b*64 + h
    int b    = blk >> 6;
    int h    = blk & 63;
    int tid  = threadIdx.x;
    int lane = tid & 31;
    int wid  = tid >> 5;
    int wn   = wid & 3;
    int wm   = wid >> 2;
    int m      = lane & 15;
    int wcol0  = wn * 16;
    int pix    = h * WW + wcol0 + m;          // this lane's pixel (A rows)
    int koff   = (lane & 16) ? 16 : 0;        // A-layout K offset per half-wave

    const float* xb = x + (long)b * CI_ * HW;
    // Flat LDS aperture: low 32 bits of the flat address are the LDS address.
    unsigned ldsB_lo = (unsigned)(uintptr_t)(&ldsB[0][0]);

    v8f acc[8];
#pragma unroll
    for (int t = 0; t < 8; ++t)
#pragma unroll
        for (int r = 0; r < 8; ++r) acc[t][r] = 0.f;

    float4 sw; int4 si;

    // ---- prologue: async-stage B(0), build A-half(0) into buffer 0 ----
    async_copy64(ldsB_lo + tid * 64, (const char*)wsB + tid * 64);
    {
        int sidx = (((b * 9) << 12) + pix) << 2;      // k9 = 0
        sw = *(const float4*)(wsW + sidx);
        si = *(const int4*)(wsI + sidx);
        unsigned av[4];
#pragma unroll
        for (int j = 0; j < 4; ++j) {
            int v = wm * 4 + j;
            int base = 2 * v + koff;
            const float* p0 = xb + (long)base * HW;   // cib = 0
            av[j] = bfpack(bilin(p0, sw, si), bilin(p0 + HW, sw, si));
        }
        *(uint4*)&ldsA[0][wn][lane][wm * 4] = *(uint4*)av;
    }
    wait_async0();

#pragma unroll 1
    for (int kc = 0; kc < NCHUNK; ++kc) {
        __syncthreads();                      // buffers [kc&1] ready
        int buf  = kc & 1;
        bool more = (kc + 1) < NCHUNK;

        // ---- issue next-chunk work early (overlaps with WMMAs below) ----
        unsigned av[4];
        if (more) {
            int bk = kc + 1;
            async_copy64(ldsB_lo + ((unsigned)(bk & 1) << 14) + tid * 64,
                         (const char*)(wsB + bk * 4096) + tid * 64);
            if ((bk & 7) == 0) {
                int k9 = bk >> 3;
                int sidx = (((b * 9 + k9) << 12) + pix) << 2;
                sw = *(const float4*)(wsW + sidx);
                si = *(const int4*)(wsI + sidx);
            }
            int cib = (bk & 7) << 5;
#pragma unroll
            for (int j = 0; j < 4; ++j) {
                int v = wm * 4 + j;
                int base = 2 * v + koff;
                const float* p0 = xb + (long)(cib + base) * HW;
                av[j] = bfpack(bilin(p0, sw, si), bilin(p0 + HW, sw, si));
            }
        }

        // ---- read current A frag + all 8 B frags, then 8 WMMAs ----
        Frag16 A;
        {
            const unsigned* ap = &ldsA[buf][wn][lane][0];
            *(uint4*)&A.u[0] = *(const uint4*)(ap);
            *(uint4*)&A.u[4] = *(const uint4*)(ap + 4);
        }
        Frag16 Bf[8];
        const unsigned* lb = &ldsB[buf][(wm * 8) * 256 + lane * 8];
#pragma unroll
        for (int t = 0; t < 8; ++t) {
            *(uint4*)&Bf[t].u[0] = *(const uint4*)(lb + t * 256);
            *(uint4*)&Bf[t].u[4] = *(const uint4*)(lb + t * 256 + 4);
        }
#pragma unroll
        for (int t = 0; t < 8; ++t)
            acc[t] = __builtin_amdgcn_wmma_f32_16x16x32_bf16(
                false, A.bf, false, Bf[t].bf, (short)0, acc[t], false, false);
#if __has_builtin(__builtin_amdgcn_sched_group_barrier)
        __builtin_amdgcn_sched_group_barrier(0x100, 18, 0); // all DS reads
        __builtin_amdgcn_sched_group_barrier(0x008, 8, 0);  // then 8 WMMAs
#endif

        // ---- commit next-chunk A half; drain own async B copy ----
        if (more) {
            *(uint4*)&ldsA[buf ^ 1][wn][lane][wm * 4] = *(uint4*)av;
        }
        wait_async0();
    }

    // Epilogue: C layout -> lanes 0-15: co=lane, M=r ; lanes 16-31: co=lane-16, M=r+8
    int nloc = lane & 15;
    int prow = (lane & 16) ? 8 : 0;
#pragma unroll
    for (int t = 0; t < 8; ++t) {
        int co = (wm * 8 + t) * 16 + nloc;
        float s  = scl[co];
        float bb = bia[co];
        float* op = out + ((long)b * CO_ + co) * HW + h * WW + wcol0 + prow;
#pragma unroll
        for (int r = 0; r < 8; ++r) {
            float v = acc[t][r] * s + bb;
            op[r] = v > 0.f ? v : 0.f;
        }
    }
}

// ---------------------------------------------------------------------------
// Workspace layout (bytes):
//   wsW : 0            .. 4718592   (B*9*HW float4 weights)
//   wsI : 4718592      .. 9437184   (B*9*HW int4 indices)
//   wsB : 9437184      .. 10616832  (packed bf16 B tiles)
//   scl : 10616832     .. 10617856
//   bia : 10617856     .. 10618880
// ---------------------------------------------------------------------------
extern "C" void kernel_launch(void* const* d_in, const int* in_sizes, int n_in,
                              void* d_out, int out_size, void* d_ws, size_t ws_size,
                              hipStream_t stream) {
    const float* x     = (const float*)d_in[0];
    const float* w_off = (const float*)d_in[1];
    const float* b_off = (const float*)d_in[2];
    const float* w_dcn = (const float*)d_in[3];
    const float* b_dcn = (const float*)d_in[4];
    const float* gamma = (const float*)d_in[5];
    const float* beta  = (const float*)d_in[6];
    const float* rmean = (const float*)d_in[7];
    const float* rvar  = (const float*)d_in[8];
    float* out = (float*)d_out;

    char* ws = (char*)d_ws;
    float*    wsW = (float*)(ws);
    int*      wsI = (int*)(ws + 4718592);
    unsigned* wsB = (unsigned*)(ws + 9437184);
    float*    scl = (float*)(ws + 10616832);
    float*    bia = (float*)(ws + 10617856);

    pack_wdcn_kernel<<<144, 256, 0, stream>>>(w_dcn, wsB);
    bn_fold_kernel<<<1, 256, 0, stream>>>(b_dcn, gamma, beta, rmean, rvar, scl, bia);
    offset_kernel<<<128, 256, 0, stream>>>(x, w_off, b_off, wsW, wsI);
    dcn_wmma_kernel<<<512, 256, 0, stream>>>(x, wsW, wsI, wsB, scl, bia, out);
}
